// HAN_28269474742364
// MI455X (gfx1250) — matrix operand order repaired and checked by reading, not compiled, not run on hardware
//
#include <hip/hip_runtime.h>
#include <hip/hip_bf16.h>

// HAN forward: GAT (per metapath/head) -> semantic attention -> DistMult scores.
// N=2560, F=512, NHID=256, H=8, M=3, D=32
#define NN     2560
#define FF     512
#define NHIDD  256
#define HH     8
#define MM     3
#define DDIM   32
#define MHCNT  (MM*HH)
#define NEGBIG (-9.0e15f)

typedef __attribute__((ext_vector_type(16))) _Float16 v16h;
typedef __attribute__((ext_vector_type(8)))  float    v8f;

#define WMMA_F16(a,b,c) __builtin_amdgcn_wmma_f32_16x16x32_f16(false,(a),false,(b),(short)0,(c),false,false)

__device__ __forceinline__ float leaky05(float x){ return x > 0.f ? x : 0.5f * x; }
__device__ __forceinline__ float elu1(float x){ return x > 0.f ? x : (expf(x) - 1.f); }

// ---------------- K1: h[mh, n, d] = features[n, :] @ W[mh, :, d]  (f16 out) ----
__global__ __launch_bounds__(32) void k_project(const float* __restrict__ feat,
                                                const float* __restrict__ W,
                                                _Float16* __restrict__ h16){
  const int tile = blockIdx.x;                 // MHCNT * (NN/16)
  const int mh   = tile / (NN/16);
  const int i0   = (tile % (NN/16)) * 16;
  const int lane = threadIdx.x;
  const int col  = lane & 15;
  const int kb8  = (lane >> 4) << 3;
  const int kb16 = (lane >> 4) << 4;
  const float* Wmh = W + (size_t)mh * FF * DDIM;
  const float* ar  = feat + (size_t)(i0 + (lane & 15)) * FF;  // A-matrix row
  v8f c0 = {}, c1 = {};
  for (int k0 = 0; k0 < FF; k0 += 32){
    v16h a;
    const float* ap = ar + k0 + kb16;
#pragma unroll
    for (int t = 0; t < 8; ++t)  a[t] = (_Float16)ap[t];
#pragma unroll
    for (int t = 8; t < 16; ++t) a[t] = (_Float16)ap[t];   // K = kb16+8..kb16+15
    v16h b0, b1;
    const float* bp = Wmh + (size_t)(k0 + kb16) * DDIM + col;
#pragma unroll
    for (int t = 0; t < 16; ++t){
      b0[t] = (_Float16)bp[t * DDIM];
      b1[t] = (_Float16)bp[t * DDIM + 16];
    }
    c0 = WMMA_F16(a, b0, c0);
    c1 = WMMA_F16(a, b1, c1);
  }
  const int rbase = (lane >> 4) << 3;
#pragma unroll
  for (int r = 0; r < 8; ++r){
    const int i = i0 + rbase + r;
    _Float16* o = h16 + ((size_t)mh * NN + i) * DDIM + col;
    o[0]  = (_Float16)c0[r];
    o[16] = (_Float16)c1[r];
  }
}

// ---------------- K2: e_src/e_dst = h . a_src / a_dst ------------------------
__global__ void k_edge(const _Float16* __restrict__ h16,
                       const float* __restrict__ a_src, const float* __restrict__ a_dst,
                       float* __restrict__ e_src, float* __restrict__ e_dst){
  const int idx = blockIdx.x * blockDim.x + threadIdx.x;   // MHCNT*NN
  if (idx >= MHCNT * NN) return;
  const int mh = idx / NN;
  const _Float16* hp = h16 + (size_t)idx * DDIM;
  const float* as = a_src + mh * DDIM;
  const float* ad = a_dst + mh * DDIM;
  float s = 0.f, d = 0.f;
#pragma unroll
  for (int t = 0; t < DDIM; ++t){
    const float hv = (float)hp[t];
    s += hv * as[t];
    d += hv * ad[t];
  }
  e_src[idx] = s;
  e_dst[idx] = d;
}

// ---------------- K3: per-row softmax stats (max, 1/sum) ----------------------
__global__ void k_stats(const float* __restrict__ e_src, const float* __restrict__ e_dst,
                        const int* __restrict__ adjs,
                        float* __restrict__ row_max, float* __restrict__ row_inv){
  const int idx = blockIdx.x * blockDim.x + threadIdx.x;   // MHCNT*NN
  if (idx >= MHCNT * NN) return;
  const int mh = idx / NN;
  const int i  = idx - mh * NN;
  const int m  = mh / HH;
  const float es = e_src[idx];
  const int*   ar = adjs + ((size_t)m * NN + i) * NN;
  const float* ed = e_dst + (size_t)mh * NN;
  float mx = -3.4e38f;
  for (int j = 0; j < NN; ++j){
    const float e = (ar[j] > 0) ? leaky05(es + ed[j]) : NEGBIG;
    mx = fmaxf(mx, e);
  }
  float s = 0.f;
  for (int j = 0; j < NN; ++j){
    const float e = (ar[j] > 0) ? leaky05(es + ed[j]) : NEGBIG;
    s += expf(e - mx);
  }
  row_max[idx] = mx;
  row_inv[idx] = 1.0f / s;
}

// ---------------- K4: fused masked-softmax attention @ h, + ELU -> Z ---------
__global__ __launch_bounds__(32) void k_attn(const _Float16* __restrict__ h16,
                                             const float* __restrict__ e_src,
                                             const float* __restrict__ e_dst,
                                             const int* __restrict__ adjs,
                                             const float* __restrict__ row_max,
                                             const float* __restrict__ row_inv,
                                             float* __restrict__ Z32,
                                             _Float16* __restrict__ Z16){
  const int tile = blockIdx.x;                 // MHCNT * (NN/16)
  const int mh   = tile / (NN/16);
  const int i0   = (tile % (NN/16)) * 16;
  const int m    = mh / HH;
  const int head = mh - m * HH;
  const int lane = threadIdx.x;
  const int col  = lane & 15;
  const int kb8  = (lane >> 4) << 3;
  const int kb16 = (lane >> 4) << 4;
  const int ia   = i0 + (lane & 15);           // A-matrix row for this lane
  const float es = e_src[(size_t)mh * NN + ia];
  const float mx = row_max[(size_t)mh * NN + ia];
  const int*   ar = adjs + ((size_t)m * NN + ia) * NN;
  const float* ed = e_dst + (size_t)mh * NN;
  const _Float16* hmh = h16 + (size_t)mh * NN * DDIM;
  v8f c0 = {}, c1 = {};
  for (int j0 = 0; j0 < NN; j0 += 32){
    v16h a;
#pragma unroll
    for (int t = 0; t < 8; ++t){
      const int j = j0 + kb16 + t;
      const float e = (ar[j] > 0) ? leaky05(es + ed[j]) : NEGBIG;
      a[t] = (_Float16)expf(e - mx);           // <= 1.0, f16-safe
    }
#pragma unroll
    for (int t = 8; t < 16; ++t){
      const int j = j0 + kb16 + t;          // K = kb16+8..kb16+15
      const float e = (ar[j] > 0) ? leaky05(es + ed[j]) : NEGBIG;
      a[t] = (_Float16)expf(e - mx);
    }
    v16h b0, b1;
    const _Float16* bp = hmh + (size_t)(j0 + kb16) * DDIM + col;
#pragma unroll
    for (int t = 0; t < 16; ++t){
      b0[t] = bp[t * DDIM];
      b1[t] = bp[t * DDIM + 16];
    }
    c0 = WMMA_F16(a, b0, c0);
    c1 = WMMA_F16(a, b1, c1);
  }
  const int rbase = (lane >> 4) << 3;
#pragma unroll
  for (int r = 0; r < 8; ++r){
    const int i = i0 + rbase + r;
    const float inv = row_inv[(size_t)mh * NN + i];
    const float v0 = elu1(c0[r] * inv);
    const float v1 = elu1(c1[r] * inv);
    const size_t zi = ((size_t)m * NN + i) * NHIDD + head * DDIM + col;
    Z32[zi]      = v0;  Z16[zi]      = (_Float16)v0;
    Z32[zi + 16] = v1;  Z16[zi + 16] = (_Float16)v1;
  }
}

// ---------------- K5: w[m,n] += q . tanh(Z @ Ws + bs)  (WMMA + reduce) -------
__global__ __launch_bounds__(32) void k_sem(const _Float16* __restrict__ Z16,
                                            const float* __restrict__ Ws,
                                            const float* __restrict__ bs,
                                            const float* __restrict__ q,
                                            float* __restrict__ wbuf){
  const int tile = blockIdx.x;                 // (MM*NN/16) * (NHIDD/32)
  const int ti = tile / (NHIDD/32);
  const int tc = tile - ti * (NHIDD/32);
  const int i0 = ti * 16;
  const int c0f = tc * 32;
  const int lane = threadIdx.x;
  const int col  = lane & 15;
  const int kb8  = (lane >> 4) << 3;
  const int kb16 = (lane >> 4) << 4;
  v8f acc0 = {}, acc1 = {};
  const _Float16* arow = Z16 + (size_t)(i0 + (lane & 15)) * NHIDD;
  for (int k0 = 0; k0 < NHIDD; k0 += 32){
    v16h a;
    const _Float16* ap = arow + k0 + kb16;
#pragma unroll
    for (int t = 0; t < 8; ++t)  a[t] = ap[t];
#pragma unroll
    for (int t = 8; t < 16; ++t) a[t] = ap[t];
    v16h b0, b1;
    const float* bp = Ws + (size_t)(k0 + kb16) * NHIDD + c0f + col;
#pragma unroll
    for (int t = 0; t < 16; ++t){
      b0[t] = (_Float16)bp[t * NHIDD];
      b1[t] = (_Float16)bp[t * NHIDD + 16];
    }
    acc0 = WMMA_F16(a, b0, acc0);
    acc1 = WMMA_F16(a, b1, acc1);
  }
  const float q0 = q[c0f + col],      q1 = q[c0f + 16 + col];
  const float b0v = bs[c0f + col],    b1v = bs[c0f + 16 + col];
  const int rbase = (lane >> 4) << 3;
#pragma unroll
  for (int r = 0; r < 8; ++r){
    float part = tanhf(acc0[r] + b0v) * q0 + tanhf(acc1[r] + b1v) * q1;
#pragma unroll
    for (int off = 1; off < 16; off <<= 1)
      part += __shfl_xor(part, off, 32);       // reduce within each 16-lane half
    if (col == 0)
      atomicAdd(&wbuf[i0 + rbase + r], part);  // row index == m*NN + n
  }
}

// ---------------- K6: beta = softmax(mean_n w), penalty = sum rel^2 ----------
__global__ __launch_bounds__(256) void k_beta(const float* __restrict__ wbuf,
                                              const float* __restrict__ rel,
                                              float* __restrict__ beta,
                                              float* __restrict__ out_penalty){
  __shared__ float sm[256];
  const int t = threadIdx.x;
  float sums[MM];
  for (int m = 0; m < MM; ++m){
    float s = 0.f;
    for (int n = t; n < NN; n += 256) s += wbuf[m * NN + n];
    sm[t] = s; __syncthreads();
    for (int off = 128; off > 0; off >>= 1){ if (t < off) sm[t] += sm[t + off]; __syncthreads(); }
    sums[m] = sm[0]; __syncthreads();
  }
  const float p = rel[t] * rel[t];             // NHIDD == blockDim == 256
  sm[t] = p; __syncthreads();
  for (int off = 128; off > 0; off >>= 1){ if (t < off) sm[t] += sm[t + off]; __syncthreads(); }
  if (t == 0){
    float wm[MM];
    for (int m = 0; m < MM; ++m) wm[m] = sums[m] / (float)NN;
    float mx = fmaxf(fmaxf(wm[0], wm[1]), wm[2]);
    float e0 = expf(wm[0] - mx), e1 = expf(wm[1] - mx), e2 = expf(wm[2] - mx);
    const float inv = 1.f / (e0 + e1 + e2);
    beta[0] = e0 * inv; beta[1] = e1 * inv; beta[2] = e2 * inv;
    out_penalty[0] = sm[0];
  }
}

// ---------------- K7: Zf = sum_m beta_m Z[m]; make f16 plain + rel-scaled ----
__global__ void k_fuse(const float* __restrict__ Z32,
                       const float* __restrict__ beta,
                       const float* __restrict__ rel,
                       _Float16* __restrict__ Zf16,
                       _Float16* __restrict__ ZfR16){
  const int idx = blockIdx.x * blockDim.x + threadIdx.x;   // NN*NHIDD
  if (idx >= NN * NHIDD) return;
  const int c = idx & (NHIDD - 1);
  const float zf = beta[0] * Z32[idx]
                 + beta[1] * Z32[(size_t)NN * NHIDD + idx]
                 + beta[2] * Z32[2 * (size_t)NN * NHIDD + idx];
  Zf16[idx]  = (_Float16)zf;
  ZfR16[idx] = (_Float16)(zf * rel[c]);
}

// ---------------- K8: scores[i,j] = (Zf_i * rel) . Zf_j  (WMMA) --------------
__global__ __launch_bounds__(32) void k_scores(const _Float16* __restrict__ ZfR,
                                               const _Float16* __restrict__ Zf,
                                               float* __restrict__ out){
  const int tile = blockIdx.x;                 // (NN/16)*(NN/32)
  const int ti = tile / (NN/32);
  const int tj = tile - ti * (NN/32);
  const int i0 = ti * 16, j0 = tj * 32;
  const int lane = threadIdx.x;
  const int col  = lane & 15;
  const int kb8  = (lane >> 4) << 3;
  const int kb16 = (lane >> 4) << 4;
  v8f c0 = {}, c1 = {};
  const _Float16* arow = ZfR + (size_t)(i0 + (lane & 15)) * NHIDD;
  const _Float16* br0  = Zf + (size_t)(j0 + col) * NHIDD;        // B col = Zf row j
  const _Float16* br1  = Zf + (size_t)(j0 + 16 + col) * NHIDD;
  for (int k0 = 0; k0 < NHIDD; k0 += 32){
    v16h a;
    const _Float16* ap = arow + k0 + kb16;
#pragma unroll
    for (int t = 0; t < 8; ++t)  a[t] = ap[t];
#pragma unroll
    for (int t = 8; t < 16; ++t) a[t] = ap[t];
    v16h b0, b1;
    const _Float16* bp0 = br0 + k0 + kb16;
    const _Float16* bp1 = br1 + k0 + kb16;
#pragma unroll
    for (int t = 0; t < 16; ++t){ b0[t] = bp0[t]; b1[t] = bp1[t]; }
    c0 = WMMA_F16(a, b0, c0);
    c1 = WMMA_F16(a, b1, c1);
  }
  const int rbase = (lane >> 4) << 3;
#pragma unroll
  for (int r = 0; r < 8; ++r){
    const size_t base = (size_t)(i0 + rbase + r) * NN;
    out[base + j0 + col]      = c0[r];
    out[base + j0 + 16 + col] = c1[r];
  }
}

// ---------------- launcher ----------------------------------------------------
extern "C" void kernel_launch(void* const* d_in, const int* in_sizes, int n_in,
                              void* d_out, int out_size, void* d_ws, size_t ws_size,
                              hipStream_t stream){
  (void)in_sizes; (void)n_in; (void)out_size; (void)ws_size;
  const float* feat  = (const float*)d_in[0];  // [N,F]
  const int*   adjs  = (const int*)d_in[1];    // [M,N,N]
  const float* W     = (const float*)d_in[2];  // [M,H,F,D]
  const float* a_src = (const float*)d_in[3];  // [M,H,D]
  const float* a_dst = (const float*)d_in[4];  // [M,H,D]
  const float* Ws    = (const float*)d_in[5];  // [NHID,NHID]
  const float* bs    = (const float*)d_in[6];  // [NHID]
  const float* q     = (const float*)d_in[7];  // [NHID]
  const float* rel   = (const float*)d_in[8];  // [NHID]
  float* out = (float*)d_out;                  // [N*N] scores + [1] penalty

  char* ws = (char*)d_ws;
  _Float16* h16    = (_Float16*)(ws + 0);          // 24*2560*32 f16   = 3,932,160 B
  float*    e_src  = (float*)(ws + 3932160);       // 61440 f32        =   245,760 B
  float*    e_dst  = (float*)(ws + 4177920);
  float*    row_mx = (float*)(ws + 4423680);
  float*    row_iv = (float*)(ws + 4669440);
  float*    Z32    = (float*)(ws + 4915200);       // 3*2560*256 f32   = 7,864,320 B
  _Float16* Z16    = (_Float16*)(ws + 12779520);   //                  = 3,932,160 B
  float*    wbuf   = (float*)(ws + 16711680);      // 7680 f32         =    30,720 B
  float*    beta   = (float*)(ws + 16742400);      // 3 f32 (padded)
  _Float16* Zf16   = (_Float16*)(ws + 16742656);   // 2560*256 f16     = 1,310,720 B
  _Float16* ZfR16  = (_Float16*)(ws + 18053376);   //                  = 1,310,720 B
  // total workspace used: 19,364,096 bytes

  k_project<<<MHCNT * (NN/16), 32, 0, stream>>>(feat, W, h16);
  k_edge<<<(MHCNT * NN + 255) / 256, 256, 0, stream>>>(h16, a_src, a_dst, e_src, e_dst);
  k_stats<<<(MHCNT * NN + 255) / 256, 256, 0, stream>>>(e_src, e_dst, adjs, row_mx, row_iv);
  k_attn<<<MHCNT * (NN/16), 32, 0, stream>>>(h16, e_src, e_dst, adjs, row_mx, row_iv, Z32, Z16);
  hipMemsetAsync(wbuf, 0, (size_t)MM * NN * sizeof(float), stream);
  k_sem<<<(MM * NN / 16) * (NHIDD / 32), 32, 0, stream>>>(Z16, Ws, bs, q, wbuf);
  k_beta<<<1, 256, 0, stream>>>(wbuf, rel, beta, out + (size_t)NN * NN);
  k_fuse<<<(NN * NHIDD + 255) / 256, 256, 0, stream>>>(Z32, beta, rel, Zf16, ZfR16);
  k_scores<<<(NN/16) * (NN/32), 32, 0, stream>>>(ZfR16, Zf16, out);
}
